// NonLinODENet_15212774162827
// MI455X (gfx1250) — matrix operand order repaired and checked by reading, not compiled, not run on hardware
//
#include <hip/hip_runtime.h>
#include <hip/hip_bf16.h>

typedef __attribute__((ext_vector_type(16))) _Float16 v16h;
typedef __attribute__((ext_vector_type(2)))  __fp16   pkh2;   // return type of cvt_pkrtz
typedef __attribute__((ext_vector_type(8)))  float    v8f;

#define SWAPX16_PATTERN 0x401F  // ds_swizzle: xor=0x10, or=0, and=0x1F (swap lanes L <-> L+16)

__device__ __forceinline__ float lane_swap16(float x) {
  return __builtin_bit_cast(float,
      __builtin_amdgcn_ds_swizzle(__builtin_bit_cast(int, x), SWAPX16_PATTERN));
}

__device__ __forceinline__ float fast_tanh(float x) {
#if __has_builtin(__builtin_amdgcn_tanhf)
  return __builtin_amdgcn_tanhf(x);          // v_tanh_f32 (gfx1250 TRANS op, co-executes)
#else
  float e = __builtin_amdgcn_exp2f(2.88539008177793f * x);
  return 1.0f - 2.0f * __builtin_amdgcn_rcpf(e + 1.0f);
#endif
}

// One RK4 stage for a 16-row batch tile held by this wave.
// State layout per lane (32 floats): index i = 8*g + v  <->  feature 16*g + hoff + v
// of batch row (lane & 15), hoff = (lane<16 ? 0 : 8). This is exactly the WMMA
// f32 C/D layout across the 4 feature tiles, so tanh/RK4 updates are elementwise.
template <bool INIT>
__device__ __forceinline__ void rk_stage(
    const v16h (&Aop)[4][2], const v8f (&bc)[4],
    const float (&sin_)[32], const float (&y)[32],
    float (&acc)[32], float (&sout)[32],
    float wacc, float cst, bool low)
{
  // Pack stage input to f16 pairs (feature 2i, 2i+1 of this lane's feature set).
  float pk[16];
#pragma unroll
  for (int i = 0; i < 16; ++i) {
    pkh2 p = __builtin_amdgcn_cvt_pkrtz(sin_[2 * i], sin_[2 * i + 1]);
    pk[i] = __builtin_bit_cast(float, p);
  }

  // Build B-operands (Y^T, K x 16): lane pair (L, L+16) holds column = batch row L.
  // low lane needs partner's features for K=8..15 / 40..47; high lane for K=16..23 / 48..55.
  v8f b0f, b1f;
#pragma unroll
  for (int j = 0; j < 4; ++j) {
    float r0 = lane_swap16(low ? pk[4 + j] : pk[j]);
    b0f[j]     = low ? pk[j]     : r0;
    b0f[j + 4] = low ? r0        : pk[4 + j];
    float r1 = lane_swap16(low ? pk[12 + j] : pk[8 + j]);
    b1f[j]     = low ? pk[8 + j] : r1;
    b1f[j + 4] = low ? r1        : pk[12 + j];
  }
  v16h B0 = __builtin_bit_cast(v16h, b0f);
  v16h B1 = __builtin_bit_cast(v16h, b1f);

  // Z^T tiles: D = A(16x32) x Y^T(32x16) + C, two K chunks, four feature tiles.
  // Two passes of 4 mutually independent WMMAs each: the pass-2 RAW on each D
  // accumulator is >= 4 WMMA instructions away, covering the f16 WMMA->WMMA
  // hazard (5 slots) without compiler-inserted v_nops.
  v8f d[4];
#pragma unroll
  for (int t = 0; t < 4; ++t)
    d[t] = __builtin_amdgcn_wmma_f32_16x16x32_f16(
        false, Aop[t][0], false, B0, (short)0, bc[t], false, false);
#pragma unroll
  for (int t = 0; t < 4; ++t)
    d[t] = __builtin_amdgcn_wmma_f32_16x16x32_f16(
        false, Aop[t][1], false, B1, (short)0, d[t], false, false);

#pragma unroll
  for (int t = 0; t < 4; ++t) {
#pragma unroll
    for (int v = 0; v < 8; ++v) {
      float k = fast_tanh(d[t][v]);
      int i = 8 * t + v;
      acc[i]  = INIT ? (wacc * k) : (acc[i] + wacc * k);
      sout[i] = y[i] + cst * k;
    }
  }
}

__global__ __launch_bounds__(256) void nonlin_ode_rk4_wmma(
    const float* __restrict__ x, const float* __restrict__ A,
    const float* __restrict__ b, const int* __restrict__ t0p,
    const int* __restrict__ tfp, float* __restrict__ out, int ntiles)
{
  const int lane = threadIdx.x & 31;
  const int wave = threadIdx.x >> 5;
  const int tile = blockIdx.x * 8 + wave;
  if (tile >= ntiles) return;           // wave-uniform; EXEC stays all-ones for WMMA

  const bool low  = lane < 16;
  const int  hoff = low ? 0 : 8;
  const int  r    = lane & 15;

  const float dt  = ((float)tfp[0] - (float)t0p[0]) * (1.0f / 40.0f);
  const float hdt = 0.5f * dt;
  const float dt6 = dt * (1.0f / 6.0f);

  // Weight tiles, f16, resident in registers for the whole integration.
  v16h Aop[4][2];
#pragma unroll
  for (int mt = 0; mt < 4; ++mt) {
#pragma unroll
    for (int kt = 0; kt < 2; ++kt) {
      const float* ap = A + (size_t)(16 * mt + r) * 64 + (32 * kt + 2 * hoff);
#pragma unroll
      for (int h = 0; h < 8; ++h) {
        Aop[mt][kt][h]     = (_Float16)ap[h];
        Aop[mt][kt][8 + h] = (_Float16)ap[8 + h];
      }
    }
  }

  // Bias tiles in the f32 C/D layout.
  v8f bc[4];
#pragma unroll
  for (int t = 0; t < 4; ++t)
#pragma unroll
    for (int v = 0; v < 8; ++v) bc[t][v] = b[16 * t + hoff + v];

  // Load initial state (row-major 524288x64) into the register layout.
  float y[32], acc[32], s[32];
  const size_t rowbase = (size_t)(tile * 16 + r) * 64;
#pragma unroll
  for (int g = 0; g < 4; ++g) {
    const float4* p = reinterpret_cast<const float4*>(x + rowbase + 16 * g + hoff);
    float4 a0 = p[0], a1 = p[1];
    y[8 * g + 0] = a0.x; y[8 * g + 1] = a0.y; y[8 * g + 2] = a0.z; y[8 * g + 3] = a0.w;
    y[8 * g + 4] = a1.x; y[8 * g + 5] = a1.y; y[8 * g + 6] = a1.z; y[8 * g + 7] = a1.w;
  }

  for (int step = 0; step < 40; ++step) {
    rk_stage<true >(Aop, bc, y, y, acc, s, 1.0f, hdt,  low);  // k1
    rk_stage<false>(Aop, bc, s, y, acc, s, 2.0f, hdt,  low);  // k2
    rk_stage<false>(Aop, bc, s, y, acc, s, 2.0f, dt,   low);  // k3
    rk_stage<false>(Aop, bc, s, y, acc, s, 1.0f, 0.0f, low);  // k4
#pragma unroll
    for (int i = 0; i < 32; ++i) y[i] += dt6 * acc[i];
  }

#pragma unroll
  for (int g = 0; g < 4; ++g) {
    float4 a0 = {y[8 * g + 0], y[8 * g + 1], y[8 * g + 2], y[8 * g + 3]};
    float4 a1 = {y[8 * g + 4], y[8 * g + 5], y[8 * g + 6], y[8 * g + 7]};
    float4* p = reinterpret_cast<float4*>(out + rowbase + 16 * g + hoff);
    p[0] = a0; p[1] = a1;
  }
}

extern "C" void kernel_launch(void* const* d_in, const int* in_sizes, int n_in,
                              void* d_out, int out_size, void* d_ws, size_t ws_size,
                              hipStream_t stream) {
  (void)n_in; (void)out_size; (void)d_ws; (void)ws_size;
  const float* x  = (const float*)d_in[0];
  const float* A  = (const float*)d_in[1];
  const float* b  = (const float*)d_in[2];
  const int*   t0 = (const int*)d_in[3];
  const int*   tf = (const int*)d_in[4];
  float* out = (float*)d_out;

  const int nrows  = in_sizes[0] / 64;      // 524288
  const int ntiles = nrows / 16;            // 32768 waves, one 16-row tile each
  const int blocks = (ntiles + 7) / 8;      // 8 waves (256 threads) per block
  nonlin_ode_rk4_wmma<<<blocks, 256, 0, stream>>>(x, A, b, t0, tf, out, ntiles);
}
